// Decoder_57320633532541
// MI455X (gfx1250) — compile-verified, matrix operand order checked by reading, not run on hardware
//
#include <hip/hip_runtime.h>
#include <math.h>

typedef __attribute__((ext_vector_type(16))) _Float16 v16h;
typedef __attribute__((ext_vector_type(8)))  _Float16 v8h;
typedef __attribute__((ext_vector_type(4)))  _Float16 h4;
typedef __attribute__((ext_vector_type(8)))  float    v8f;

#define DEVFN static __device__ __forceinline__

constexpr int BATCH = 4;
constexpr int NNODE = 4096;
constexpr int HID   = 128;
constexpr int G4    = 512;   // 4*HID LSTM gate width

// ---------------- workspace layout (bytes, 256-aligned) ----------------
constexpr size_t OFF_DINV = 0;                        // float [4*4096]
constexpr size_t OFF_YT   = OFF_DINV + 65536;         // fp16  [4][128][4096]  (yT[b][n][k] = d_inv[k]*x[k][n])
constexpr size_t OFF_DYN  = OFF_YT   + 4194304;       // float [4][128][128]   conv output
constexpr size_t OFF_XP   = OFF_DYN  + 262144;        // float [128][4][512]   X_proj + biases
constexpr size_t OFF_LWH  = OFF_XP   + 1048576;       // fp16  [4][128][128]   LSTM hidden sequence
constexpr size_t OFF_WHH  = OFF_LWH  + 131072;        // fp16  [512][128]      w_hh fp16
constexpr size_t OFF_AGG  = OFF_WHH  + 131072;        // float [4][4096][128]  agg
// total = 14,221,312 bytes

DEVFN v16h ldop(const _Float16* p) {
  // Gather one WMMA 16-bit operand: lanes 0-15 take K[0..7]+K[16..23],
  // lanes 16-31 take K[8..15]+K[24..31]; caller supplies p already offset
  // by (lane>>4)*8 within the row.  Two 16B loads -> ds_load_b128.
  v8h lo = *(const v8h*)p;
  v8h hi = *(const v8h*)(p + 16);
  v16h r;
#pragma unroll
  for (int i = 0; i < 8; ++i) { r[i] = lo[i]; r[i + 8] = hi[i]; }
  return r;
}

DEVFN float sigmoidf(float x) { return 1.0f / (1.0f + __expf(-x)); }

// ---------------------------------------------------------------------------
// K1: d_inv[b*N+i] = rsqrt(1 + rowsum(A[b,i,:]))   (one wave per row)
// ---------------------------------------------------------------------------
__global__ void k_rowsum(const float* __restrict__ A, float* __restrict__ dinv) {
  int wave = threadIdx.x >> 5, lane = threadIdx.x & 31;
  int R = blockIdx.x * 8 + wave;                 // 0..16383  ( = b*4096 + i )
  const float* row = A + (size_t)R * NNODE;
  float s = 0.0f;
#pragma unroll 4
  for (int k = lane * 4; k < NNODE; k += 128) {
    float4 v = *(const float4*)(row + k);
    s += v.x + v.y + v.z + v.w;
  }
#pragma unroll
  for (int off = 16; off; off >>= 1) s += __shfl_xor(s, off, 32);
  if (lane == 0) dinv[R] = rsqrtf(s + 1.0f);     // A_hat = A + I  =>  d >= 1
}

// ---------------------------------------------------------------------------
// K2: w_hh -> fp16
// ---------------------------------------------------------------------------
__global__ void k_cvt_whh(const float* __restrict__ w, _Float16* __restrict__ wh) {
  int i = blockIdx.x * 256 + threadIdx.x;
  if (i < G4 * HID) wh[i] = (_Float16)w[i];
}

// ---------------------------------------------------------------------------
// K3: yT[b][n][k] = (fp16)(d_inv[b*N+k] * x[b,k,n])   tiled 32x32 transpose
// ---------------------------------------------------------------------------
__global__ void k_scale_transpose(const float* __restrict__ x,
                                  const float* __restrict__ dinv,
                                  _Float16* __restrict__ yT) {
  __shared__ float tile[32][33];
  int j0 = blockIdx.x * 32, n0 = blockIdx.y * 32, b = blockIdx.z;
  int tid = threadIdx.x;
#pragma unroll
  for (int e = tid; e < 1024; e += 256) {
    int r = e >> 5, c = e & 31;
    tile[r][c] = dinv[b * NNODE + j0 + r] *
                 x[(size_t)(b * NNODE + j0 + r) * HID + n0 + c];
  }
  __syncthreads();
#pragma unroll
  for (int e = tid; e < 1024; e += 256) {
    int rn = e >> 5, cj = e & 31;
    yT[(size_t)(b * HID + n0 + rn) * NNODE + j0 + cj] = (_Float16)tile[cj][rn];
  }
}

// ---------------------------------------------------------------------------
// K4: conv1d (NCH, OIH, pad 1) on df_in = x[:, 3968:4096, :]
//     dyn[b,o,l] = conv_b[o] + sum_{c,kk} df[b,c,l+kk-1] * conv_w[o,c,kk]
// ---------------------------------------------------------------------------
__global__ void k_conv(const float* __restrict__ x, const float* __restrict__ cw,
                       const float* __restrict__ cb, float* __restrict__ dyn) {
  __shared__ float df[64 * 128];                 // 32 KB: 64 channels per phase
  int oc = blockIdx.x, b = blockIdx.y;
  int tid = threadIdx.x;
  int l = tid & 127, oo = tid >> 7;              // each thread: 4 outputs
  int obase = oc * 8 + oo;
  float acc[4];
#pragma unroll
  for (int q = 0; q < 4; ++q) acc[q] = cb[obase + 2 * q];

  for (int phase = 0; phase < 2; ++phase) {
    for (int e = tid; e < 8192; e += 256) {
      int cc = e >> 7, ll = e & 127;
      df[e] = x[(size_t)(b * NNODE + (NNODE - HID) + phase * 64 + cc) * HID + ll];
    }
    __syncthreads();
    for (int cc = 0; cc < 64; ++cc) {
      int c = phase * 64 + cc;
      float xm = (l > 0)   ? df[cc * 128 + l - 1] : 0.0f;
      float x0 =             df[cc * 128 + l];
      float xp = (l < 127) ? df[cc * 128 + l + 1] : 0.0f;
#pragma unroll
      for (int q = 0; q < 4; ++q) {
        const float* wv = cw + ((obase + 2 * q) * 128 + c) * 3;
        acc[q] = fmaf(xm, wv[0], fmaf(x0, wv[1], fmaf(xp, wv[2], acc[q])));
      }
    }
    __syncthreads();
  }
#pragma unroll
  for (int q = 0; q < 4; ++q)
    dyn[(size_t)(b * HID + obase + 2 * q) * HID + l] = acc[q];
}

// ---------------------------------------------------------------------------
// K5: Xp[t][b][g] = b_ih[g] + b_hh[g] + sum_l dyn[b,t,l] * w_ih[g,l]
// ---------------------------------------------------------------------------
__global__ void k_xproj(const float* __restrict__ dyn, const float* __restrict__ wih,
                        const float* __restrict__ bih, const float* __restrict__ bhh,
                        float* __restrict__ Xp) {
  __shared__ float xt[128];
  int t = blockIdx.x, b = blockIdx.y, tid = threadIdx.x;
  if (tid < 128) xt[tid] = dyn[(size_t)(b * HID + t) * HID + tid];
  __syncthreads();
  for (int g = tid; g < G4; g += 256) {
    float s = bih[g] + bhh[g];
    const float* wr = wih + (size_t)g * HID;
#pragma unroll 8
    for (int l = 0; l < HID; ++l) s = fmaf(xt[l], wr[l], s);
    Xp[(size_t)(t * BATCH + b) * G4 + g] = s;
  }
}

// ---------------------------------------------------------------------------
// K6: sequential LSTM, single workgroup, 512 threads = 16 waves.
//     Each wave owns 32 gate columns; recurrent weights live in 8 v16h
//     (64 VGPRs) per wave -> no scratch spills.  Per step: 8 WMMAs/wave.
//     z_t = Xp[t] + h @ w_hh.T  ->  gates  ->  h,c update; h -> lw (fp16).
// ---------------------------------------------------------------------------
__global__ void __launch_bounds__(512, 1)
k_lstm(const float* __restrict__ Xp, const _Float16* __restrict__ whh,
       _Float16* __restrict__ lw) {
  constexpr int HSTR = 136;                       // padded K-stride for hA
  __shared__ _Float16 hA[16 * HSTR];              // A operand: rows 0..3 = batch
  __shared__ float    cst[BATCH * HID];
  __shared__ float    zbuf[BATCH * G4];

  int tid = threadIdx.x;
  int wave = tid >> 5, lane = tid & 31;
  int laneM = lane & 15, kh = (lane >> 4) * 8;

  // preload this wave's recurrent-weight B-operands (g range [wave*32, +32))
  v16h bw[8];
#pragma unroll
  for (int nb = 0; nb < 2; ++nb)
#pragma unroll
    for (int kt = 0; kt < 4; ++kt) {
      int g = wave * 32 + nb * 16 + laneM;
      bw[nb * 4 + kt] = ldop(whh + (size_t)g * HID + kt * 32 + kh);
    }

  for (int e = tid; e < 16 * HSTR; e += 512) hA[e] = (_Float16)0.0f;
  if (tid < BATCH * HID) cst[tid] = 0.0f;
  __syncthreads();

  for (int t = 0; t < HID; ++t) {
    v16h aOp[4];
#pragma unroll
    for (int kt = 0; kt < 4; ++kt)
      aOp[kt] = ldop(&hA[laneM * HSTR + kt * 32 + kh]);

#pragma unroll
    for (int nb = 0; nb < 2; ++nb) {
      v8f acc = {};
#pragma unroll
      for (int kt = 0; kt < 4; ++kt)
        acc = __builtin_amdgcn_wmma_f32_16x16x32_f16(
            false, aOp[kt], false, bw[nb * 4 + kt], (short)0, acc, false, false);
      if (lane < 16) {                            // rows m=0..3 hold batch
#pragma unroll
        for (int r = 0; r < 4; ++r)
          zbuf[r * G4 + wave * 32 + nb * 16 + laneM] = acc[r];
      }
    }
    __syncthreads();

    // gate math: 512 cells (b,j), one per thread
    {
      int bb = tid >> 7, j = tid & 127;
      const float* xp = Xp + (size_t)(t * BATCH + bb) * G4;
      float zi = zbuf[bb * G4 + j]           + xp[j];
      float zf = zbuf[bb * G4 + HID + j]     + xp[HID + j];
      float zg = zbuf[bb * G4 + 2 * HID + j] + xp[2 * HID + j];
      float zo = zbuf[bb * G4 + 3 * HID + j] + xp[3 * HID + j];
      float c  = sigmoidf(zf) * cst[tid] + sigmoidf(zi) * tanhf(zg);
      float h  = sigmoidf(zo) * tanhf(c);
      cst[tid] = c;
      hA[bb * HSTR + j] = (_Float16)h;
      lw[(size_t)(bb * HID + t) * HID + j] = (_Float16)h;
    }
    __syncthreads();
  }
}

// ---------------------------------------------------------------------------
// K7: agg = d_inv[i]*(A @ yT') + d_inv[i]^2 * x     (the 17.2 GFLOP matmul)
//     WG tile: 64 rows x 128 cols, K loop over 4096 in steps of 64.
// ---------------------------------------------------------------------------
__global__ void k_aggmm(const float* __restrict__ A, const _Float16* __restrict__ yT,
                        const float* __restrict__ dinv, const float* __restrict__ x,
                        float* __restrict__ agg) {
  constexpr int ASTR = 72, BSTR = 72;             // padded fp16 strides (144 B)
  __shared__ _Float16 Ah[64 * ASTR];              // 9  KB
  __shared__ _Float16 Bh[128 * BSTR];             // 18 KB
  int b = blockIdx.y, i0 = blockIdx.x * 64;
  int tid = threadIdx.x;
  int wave = tid >> 5, lane = tid & 31;
  int mb = wave >> 1, nbs = (wave & 1) * 4;
  int laneM = lane & 15, kh = (lane >> 4) * 8;

  v8f acc[4] = {};

  for (int k0 = 0; k0 < NNODE; k0 += 64) {
    // stage A (64x64 fp32 -> fp16); A entries are {0,1}: conversion exact
#pragma unroll
    for (int c = tid; c < 1024; c += 256) {
      int row = c >> 4, q = c & 15;
      float4 f = *(const float4*)(A + (size_t)(b * NNODE + i0 + row) * NNODE + k0 + q * 4);
      h4 hv = {(_Float16)f.x, (_Float16)f.y, (_Float16)f.z, (_Float16)f.w};
      *(h4*)&Ah[row * ASTR + q * 4] = hv;
    }
    // stage B (yT rows: [N=128][K=64]) - contiguous 16B copies
#pragma unroll
    for (int c = tid; c < 1024; c += 256) {
      int n = c >> 3, q = c & 7;
      *(v8h*)&Bh[n * BSTR + q * 8] =
          *(const v8h*)(yT + (size_t)(b * HID + n) * NNODE + k0 + q * 8);
    }
    if (k0 + 64 < NNODE)                          // prefetch next A tile
      __builtin_prefetch(A + (size_t)(b * NNODE + i0 + (tid >> 2)) * NNODE + k0 + 64, 0, 1);
    __syncthreads();

#pragma unroll
    for (int kk = 0; kk < 64; kk += 32) {
      v16h aOp = ldop(&Ah[(mb * 16 + laneM) * ASTR + kk + kh]);
#pragma unroll
      for (int j = 0; j < 4; ++j) {
        v16h bOp = ldop(&Bh[((nbs + j) * 16 + laneM) * BSTR + kk + kh]);
        acc[j] = __builtin_amdgcn_wmma_f32_16x16x32_f16(
            false, aOp, false, bOp, (short)0, acc[j], false, false);
      }
    }
    __syncthreads();
  }

  // epilogue: C layout -> lane holds n=lane&15, rows m=(lane>>4)*8 + r
#pragma unroll
  for (int r = 0; r < 8; ++r) {
    int i = i0 + mb * 16 + (lane >> 4) * 8 + r;
    float di = dinv[b * NNODE + i];
#pragma unroll
    for (int j = 0; j < 4; ++j) {
      int n = (nbs + j) * 16 + laneM;
      float xv = x[(size_t)(b * NNODE + i) * HID + n];
      agg[(size_t)(b * NNODE + i) * HID + n] = di * acc[j][r] + di * di * xv;
    }
  }
}

// ---------------------------------------------------------------------------
// K8: out = sigmoid(agg @ lw)     M=4096 (128/WG), N=128, K=128
// ---------------------------------------------------------------------------
__global__ void k_outmm(const float* __restrict__ agg, const _Float16* __restrict__ lw,
                        float* __restrict__ out) {
  constexpr int LSTR = 136, ASTR = 40;
  __shared__ _Float16 lwT[128 * LSTR];            // 34 KB  [N=h][K=t]
  __shared__ _Float16 Ah[128 * ASTR];             // 10 KB
  int b = blockIdx.y, i0 = blockIdx.x * 128;
  int tid = threadIdx.x;
  int wave = tid >> 5, lane = tid & 31;
  int laneM = lane & 15, kh = (lane >> 4) * 8;

  for (int e = tid; e < HID * HID; e += 256) {
    int t = e >> 7, h = e & 127;
    lwT[h * LSTR + t] = lw[(size_t)(b * HID + t) * HID + h];
  }

  v8f acc[8] = {};
  for (int kt = 0; kt < 4; ++kt) {
    __syncthreads();
#pragma unroll
    for (int c = tid; c < 1024; c += 256) {
      int row = c >> 3, q = c & 7;
      const float* s = agg + (size_t)(b * NNODE + i0 + row) * HID + kt * 32 + q * 4;
      h4 hv = {(_Float16)s[0], (_Float16)s[1], (_Float16)s[2], (_Float16)s[3]};
      *(h4*)&Ah[row * ASTR + q * 4] = hv;
    }
    __syncthreads();

    v16h aOp = ldop(&Ah[(wave * 16 + laneM) * ASTR + kh]);
#pragma unroll
    for (int nb = 0; nb < 8; ++nb) {
      v16h bOp = ldop(&lwT[(nb * 16 + laneM) * LSTR + kt * 32 + kh]);
      acc[nb] = __builtin_amdgcn_wmma_f32_16x16x32_f16(
          false, aOp, false, bOp, (short)0, acc[nb], false, false);
    }
  }

#pragma unroll
  for (int r = 0; r < 8; ++r) {
    int i = i0 + wave * 16 + (lane >> 4) * 8 + r;
#pragma unroll
    for (int nb = 0; nb < 8; ++nb) {
      int n = nb * 16 + laneM;
      out[(size_t)(b * NNODE + i) * HID + n] = sigmoidf(acc[nb][r]);
    }
  }
}

// ---------------------------------------------------------------------------
extern "C" void kernel_launch(void* const* d_in, const int* in_sizes, int n_in,
                              void* d_out, int out_size, void* d_ws, size_t ws_size,
                              hipStream_t stream) {
  const float* x      = (const float*)d_in[0];
  const float* A      = (const float*)d_in[1];
  const float* conv_w = (const float*)d_in[2];
  const float* conv_b = (const float*)d_in[3];
  const float* w_ih   = (const float*)d_in[4];
  const float* w_hh   = (const float*)d_in[5];
  const float* b_ih   = (const float*)d_in[6];
  const float* b_hh   = (const float*)d_in[7];
  float* out = (float*)d_out;

  char* ws = (char*)d_ws;
  float*    dinv  = (float*)(ws + OFF_DINV);
  _Float16* yT    = (_Float16*)(ws + OFF_YT);
  float*    dyn   = (float*)(ws + OFF_DYN);
  float*    Xp    = (float*)(ws + OFF_XP);
  _Float16* lw    = (_Float16*)(ws + OFF_LWH);
  _Float16* whh   = (_Float16*)(ws + OFF_WHH);
  float*    agg   = (float*)(ws + OFF_AGG);

  k_rowsum<<<2048, 256, 0, stream>>>(A, dinv);
  k_cvt_whh<<<256, 256, 0, stream>>>(w_hh, whh);
  k_scale_transpose<<<dim3(128, 4, 4), 256, 0, stream>>>(x, dinv, yT);
  k_conv<<<dim3(16, 4), 256, 0, stream>>>(x, conv_w, conv_b, dyn);
  k_xproj<<<dim3(128, 4), 256, 0, stream>>>(dyn, w_ih, b_ih, b_hh, Xp);
  k_lstm<<<1, 512, 0, stream>>>(Xp, whh, lw);
  k_aggmm<<<dim3(64, 4), 256, 0, stream>>>(A, yT, dinv, x, agg);
  k_outmm<<<dim3(32, 4), 256, 0, stream>>>(agg, lw, out);
}
